// PositionAttention3D_20607253086546
// MI455X (gfx1250) — matrix operand order repaired and checked by reading, not compiled, not run on hardware
//
#include <hip/hip_runtime.h>

// ---------------------------------------------------------------------------
// PositionAttention3D for MI455X (gfx1250, wave32, WMMA)
//
// B=2, C=64, INTER=32, N = 16*24*24 = 9216. ~33 GFLOP total; x is 4.7MB
// (L2-resident). Streaming flash-attention avoids the 680MB energy tensor
// and is compute-bound on v_wmma_f32_16x16x32_bf16.
//
// Key layout trick: S^T = K * Q^T puts each query in one lane column, so the
// post-softmax P values sit *already* in the A-operand 16x32 layout for the
// second matmul O = P * V -- no cross-lane transpose needed at all.
// ---------------------------------------------------------------------------

typedef __attribute__((ext_vector_type(16))) __bf16 v16bf;
typedef __attribute__((ext_vector_type(8)))  __bf16 v8bf;
typedef __attribute__((ext_vector_type(8)))  float  v8f;

#define BB    2
#define CC    64
#define NN    9216      // 16*24*24
#define NI    32        // INTER = C/2

// ---------------------------------------------------------------------------
// Phase 0: QKV projections (tiny: ~0.3 GFLOP). One thread per output element.
//   Q,K stored as [B, N, 32] row-major bf16 (rows 64B-aligned)
//   V   stored as [B, C, N]  channel-major bf16 (rows 64B-aligned)
// ---------------------------------------------------------------------------
__global__ __launch_bounds__(256) void pa3d_qkv_kernel(
    const float* __restrict__ x,
    const float* __restrict__ Wq, const float* __restrict__ bq,
    const float* __restrict__ Wk, const float* __restrict__ bk,
    const float* __restrict__ Wv, const float* __restrict__ bv,
    __bf16* __restrict__ Qb, __bf16* __restrict__ Kb, __bf16* __restrict__ Vb)
{
    const int ROWS = NI + NI + CC;                 // 128 output rows per batch
    int tid = blockIdx.x * blockDim.x + threadIdx.x;
    if (tid >= BB * ROWS * NN) return;

    int n = tid % NN;
    int r = (tid / NN) % ROWS;
    int b = tid / (NN * ROWS);

    const float* xb = x + (size_t)b * CC * NN + n;   // x[b, :, n], stride N
    const float* wrow;
    float acc;
    if (r < NI)            { wrow = Wq + r * CC;            acc = bq[r]; }
    else if (r < 2 * NI)   { wrow = Wk + (r - NI) * CC;     acc = bk[r - NI]; }
    else                   { wrow = Wv + (r - 2 * NI) * CC; acc = bv[r - 2 * NI]; }

#pragma unroll 8
    for (int c = 0; c < CC; ++c)
        acc = fmaf(wrow[c], xb[(size_t)c * NN], acc);

    if (r < NI)          Qb[((size_t)b * NN + n) * NI + r]            = (__bf16)acc;
    else if (r < 2 * NI) Kb[((size_t)b * NN + n) * NI + (r - NI)]     = (__bf16)acc;
    else                 Vb[((size_t)b * CC + (r - 2 * NI)) * NN + n] = (__bf16)acc;
}

// ---------------------------------------------------------------------------
// Phase 1: flash attention. 256 threads = 8 waves; each wave owns 16 queries
// and streams keys in 64-key steps (144 iterations):
//   S^T(16k x 16q) = K_tile(16x32) * Q^T(32x16)   -> 4x v_wmma per step
//   online softmax: per-query stats live per-lane (query = lane&15); row
//     reduction = in-lane max/sum + one xor-16 combine
//   O(16q x 16c) += P(16x32) * V(32x16)           -> 8x v_wmma per step
//     P's A-operand IS the exp'd S^T output (no transpose); V B-operand is
//     two 16B loads (k = 8h+i, 16+8h+i) from the [C,N] layout.
// Per-row alpha for the O rescale: 8 ds_bpermute broadcasts per step.
// No divergence -> EXEC all ones for every WMMA (ISA 7.12 requirement).
// ---------------------------------------------------------------------------
__global__ __launch_bounds__(256) void pa3d_attn_kernel(
    const __bf16* __restrict__ Qb, const __bf16* __restrict__ Kb,
    const __bf16* __restrict__ Vb,
    const float* __restrict__ x, const float* __restrict__ gamma,
    float* __restrict__ out)
{
    const int lane = threadIdx.x & 31;
    const int wv   = threadIdx.x >> 5;
    const int h    = lane >> 4;          // half of the wave
    const int m    = lane & 15;          // A-row / B,C,D column index

    const int tilesPerBatch = NN / 16;   // 576
    int gt = blockIdx.x * 8 + wv;        // global 16-query tile id
    int b  = gt / tilesPerBatch;
    int q0 = (gt % tilesPerBatch) * 16;

    // Loop-invariant B operand for S^T: Q^T (K-dim = 32 features). 32B load.
    v16bf bQ = *(const v16bf*)(Qb + ((size_t)b * NN + q0 + m) * NI + h * 16);

    const __bf16* Kbase = Kb + (size_t)b * NN * NI;
    const __bf16* Vbase = Vb + (size_t)b * CC * NN;

    v8f accv[4];                         // O[16q x 64c]: row = query r+h*8
#pragma unroll
    for (int t = 0; t < 4; ++t) accv[t] = (v8f){};
    float m_run = -__builtin_inff();
    float l_run = 0.f;

    for (int kt = 0; kt < NN; kt += 64) {
        // ---- S^T: four 16-key tiles -----------------------------------------
        v8f s[4];
#pragma unroll
        for (int t = 0; t < 4; ++t) {
            const __bf16* kr = Kbase + (size_t)(kt + t * 16 + m) * NI;
            v8bf alo = *(const v8bf*)(kr + h * 16);
            v8bf ahi = *(const v8bf*)(kr + h * 16 + 8);
            v16bf A;
#pragma unroll
            for (int i = 0; i < 8; ++i) { A[i] = alo[i]; A[8 + i] = ahi[i]; }
            v8f z = (v8f){};
            s[t] = __builtin_amdgcn_wmma_f32_16x16x32_bf16(
                false, A, false, bQ, (short)0, z, false, false);
        }
        // prefetch next step's K rows (global_prefetch_b8 path)
        __builtin_prefetch(Kbase + (size_t)(kt + 64 + m) * NI, 0, 1);

        // ---- online softmax over the 64 keys (per-lane query column) --------
        float mloc = s[0][0];
#pragma unroll
        for (int t = 0; t < 4; ++t)
#pragma unroll
            for (int r = 0; r < 8; ++r) mloc = fmaxf(mloc, s[t][r]);
        mloc = fmaxf(mloc, __shfl_xor(mloc, 16, 32));
        float m_new = fmaxf(m_run, mloc);
        float alpha = __expf(m_run - m_new);
        m_run = m_new;

        float lloc = 0.f;
#pragma unroll
        for (int t = 0; t < 4; ++t)
#pragma unroll
            for (int r = 0; r < 8; ++r) {
                s[t][r] = __expf(s[t][r] - m_new);
                lloc += s[t][r];
            }
        lloc += __shfl_xor(lloc, 16, 32);
        l_run = l_run * alpha + lloc;

        // ---- rescale O: per-row alpha (row = query r+h*8) -------------------
#pragma unroll
        for (int r = 0; r < 8; ++r) {
            float ar = __shfl(alpha, h * 8 + r, 32);
#pragma unroll
            for (int t = 0; t < 4; ++t) accv[t][r] *= ar;
        }

        // ---- P A-operands: exp'd S^T values are already in A 16x32 layout ---
        v16bf AP0, AP1;
#pragma unroll
        for (int i = 0; i < 8; ++i) {
            AP0[i]     = (__bf16)s[0][i];    // k = h*8+i
            AP0[8 + i] = (__bf16)s[1][i];    // k = 16+h*8+i
            AP1[i]     = (__bf16)s[2][i];
            AP1[8 + i] = (__bf16)s[3][i];
        }

        // ---- O += P * V : 4 channel chunks x 2 key halves -------------------
#pragma unroll
        for (int t = 0; t < 4; ++t) {
            const __bf16* vrow = Vbase + (size_t)(t * 16 + m) * NN + kt;
            v8bf b0lo = *(const v8bf*)(vrow + h * 8);
            v8bf b0hi = *(const v8bf*)(vrow + 16 + h * 8);
            v8bf b1lo = *(const v8bf*)(vrow + 32 + h * 8);
            v8bf b1hi = *(const v8bf*)(vrow + 48 + h * 8);
            v16bf Bv0, Bv1;
#pragma unroll
            for (int i = 0; i < 8; ++i) {
                Bv0[i] = b0lo[i]; Bv0[8 + i] = b0hi[i];
                Bv1[i] = b1lo[i]; Bv1[8 + i] = b1hi[i];
            }
            accv[t] = __builtin_amdgcn_wmma_f32_16x16x32_bf16(
                false, AP0, false, Bv0, (short)0, accv[t], false, false);
            accv[t] = __builtin_amdgcn_wmma_f32_16x16x32_bf16(
                false, AP1, false, Bv1, (short)0, accv[t], false, false);
        }
    }

    // ---- epilogue: out = gamma/l * O + x (contiguous in n across regs) -----
    float gl = gamma[0] / l_run;         // per-query scale (lane's query m)
    const float* xb = x   + (size_t)b * CC * NN;
    float*       ob = out + (size_t)b * CC * NN;
#pragma unroll
    for (int r = 0; r < 8; ++r) {
        float glr = __shfl(gl, h * 8 + r, 32);       // scale for query r+h*8
        int n = q0 + h * 8 + r;
#pragma unroll
        for (int t = 0; t < 4; ++t) {
            int ch = t * 16 + m;                     // D column = lane&15
            size_t idx = (size_t)ch * NN + n;
            ob[idx] = glr * accv[t][r] + xb[idx];
        }
    }
}

// ---------------------------------------------------------------------------
extern "C" void kernel_launch(void* const* d_in, const int* in_sizes, int n_in,
                              void* d_out, int out_size, void* d_ws, size_t ws_size,
                              hipStream_t stream)
{
    const float* x     = (const float*)d_in[0];
    const float* Wq    = (const float*)d_in[1];
    const float* bq    = (const float*)d_in[2];
    const float* Wk    = (const float*)d_in[3];
    const float* bk    = (const float*)d_in[4];
    const float* Wv    = (const float*)d_in[5];
    const float* bv    = (const float*)d_in[6];
    const float* gamma = (const float*)d_in[7];
    float* out = (float*)d_out;

    // Workspace partition (all 64B-aligned): Q,K [B,N,32] bf16; V [B,C,N] bf16.
    __bf16* Qb = (__bf16*)d_ws;
    __bf16* Kb = Qb + (size_t)BB * NN * NI;
    __bf16* Vb = Kb + (size_t)BB * NN * NI;

    int total  = BB * (2 * NI + CC) * NN;            // 2,359,296 elements
    pa3d_qkv_kernel<<<(total + 255) / 256, 256, 0, stream>>>(
        x, Wq, bq, Wk, bk, Wv, bv, Qb, Kb, Vb);

    int blocks = BB * NN / (16 * 8);                 // 144 blocks of 8 waves
    pa3d_attn_kernel<<<blocks, 256, 0, stream>>>(Qb, Kb, Vb, x, gamma, out);
}
